// TrafficPatternMemory_39187281609423
// MI455X (gfx1250) — compile-verified, matrix operand order checked by reading, not run on hardware
//
#include <hip/hip_runtime.h>
#include <math.h>

// Problem constants (from reference): M = B*T*V rows
#define MTOT   (4*24*2048)   // 196608
#define HDIM   512
#define CDIM   256
#define NDIM   512
#define WAVES  4
#define TPB    (WAVES*32)
#define ROWS_PER_BLOCK (WAVES*16)
#define BSTRIDE 520          // bf16 units per staged row (1040B) -> no 64-bank conflicts
#define SCW     516          // f32 units per score row (2064B)   -> no 64-bank conflicts

typedef __bf16 bf16;
typedef __attribute__((ext_vector_type(16))) __bf16 v16bf;
typedef __attribute__((ext_vector_type(8)))  float  v8f;
typedef __attribute__((ext_vector_type(4)))  float  v4f;

// ---------------------------------------------------------------------------
// Prep: W_s = W_p @ W_q  (N x H, bf16), W_pT (C x N, bf16), bias_s = W_p @ b
// ---------------------------------------------------------------------------
__global__ __launch_bounds__(128) void prep_kernel(
    const float* __restrict__ Wp,   // (N, C)
    const float* __restrict__ Wq,   // (C, H)
    const float* __restrict__ Wqb,  // (C)
    bf16* __restrict__ Ws,          // (N, H)
    bf16* __restrict__ WpT,         // (C, N)
    float* __restrict__ biasS)      // (N)
{
    __shared__ float wprow[CDIM];
    const int n   = blockIdx.x;
    const int tid = threadIdx.x;
    wprow[tid]       = Wp[(size_t)n * CDIM + tid];
    wprow[tid + 128] = Wp[(size_t)n * CDIM + tid + 128];
    __syncthreads();

    for (int h = tid; h < HDIM; h += 128) {
        float a = 0.f;
        for (int c = 0; c < CDIM; ++c) a += wprow[c] * Wq[(size_t)c * HDIM + h];
        Ws[(size_t)n * HDIM + h] = (bf16)a;
    }
    // transpose W_p to (C,N) bf16 so B-fragments of GEMM-2 read contiguous K
    WpT[(size_t)tid * NDIM + n]         = (bf16)wprow[tid];
    WpT[(size_t)(tid + 128) * NDIM + n] = (bf16)wprow[tid + 128];
    if (tid == 0) {
        float b = 0.f;
        for (int c = 0; c < CDIM; ++c) b += wprow[c] * Wqb[c];
        biasS[n] = b;
    }
}

// ---------------------------------------------------------------------------
// Fused: scores = Hl @ Ws^T + bias -> softmax(x10) -> Q (out), Pt = Q @ Wp
// ---------------------------------------------------------------------------
__global__ __launch_bounds__(TPB) void pq_fused_kernel(
    const float* __restrict__ Hl,   // (M, H) f32
    const bf16*  __restrict__ Ws,   // (N, H) bf16, L2-resident
    const bf16*  __restrict__ WpT,  // (C, N) bf16, L2-resident
    const float* __restrict__ biasS,// (N)
    float* __restrict__ Pt,         // (M, C)
    float* __restrict__ Qout)       // (M, N)
{
    __shared__ bf16  bstage[16 * BSTRIDE];      // 16.6 KB staged B tile
    __shared__ float sc[WAVES][16][SCW];        // 129 KB scores/Q

    const int tid  = threadIdx.x;
    const int wave = tid >> 5;
    const int lane = tid & 31;
    const int m0   = blockIdx.x * ROWS_PER_BLOCK + wave * 16;

    const int mrow = lane & 15;                 // A-fragment row / B-fragment col
    const int kb1  = (lane < 16) ? 0 : 8;       // ISA 16-bit fragment K sub-offset

    // ---- load resident A-strip: 16 rows x 512 K, f32 -> bf16 WMMA fragments
    //      (streamed once: non-temporal, keep L2 for the weight tiles)
    v16bf afrag[16];
    {
        const float* rp = Hl + (size_t)(m0 + mrow) * HDIM;
        #pragma unroll
        for (int ks = 0; ks < 16; ++ks) {
            const int k0 = ks * 32;
            const v4f* p0 = (const v4f*)(rp + k0 + kb1);        // K: kb1 .. kb1+7
            const v4f* p1 = (const v4f*)(rp + k0 + kb1 + 16);   // K: kb1+16 .. +23
            v4f a0 = __builtin_nontemporal_load(p0);
            v4f a1 = __builtin_nontemporal_load(p0 + 1);
            v4f b0 = __builtin_nontemporal_load(p1);
            v4f b1 = __builtin_nontemporal_load(p1 + 1);
            v16bf f;
            #pragma unroll
            for (int j = 0; j < 4; ++j) {
                f[j]      = (bf16)a0[j];
                f[4 + j]  = (bf16)a1[j];
                f[8 + j]  = (bf16)b0[j];
                f[12 + j] = (bf16)b1[j];
            }
            afrag[ks] = f;
        }
    }

    const int stgRow = tid >> 3;   // 0..15: staged row
    const int stgSub = tid & 7;    // 8 threads x 8 uint4 = 1KB row

    // ================= Phase A: scores over 32 column tiles =================
    for (int nt = 0; nt < NDIM / 16; ++nt) {
        __syncthreads();
        {   // cooperatively stage 16 rows of Ws (column tile of B) into LDS
            const uint4* src = (const uint4*)(Ws + ((size_t)nt * 16 + stgRow) * HDIM);
            uint4* dst = (uint4*)(bstage + stgRow * BSTRIDE);
            #pragma unroll
            for (int i = 0; i < 8; ++i) dst[stgSub + i * 8] = src[stgSub + i * 8];
        }
        if (nt + 1 < NDIM / 16)
            __builtin_prefetch(Ws + ((size_t)(nt + 1) * 16 + stgRow) * HDIM, 0, 0);
        __syncthreads();

        v8f acc = {0.f,0.f,0.f,0.f,0.f,0.f,0.f,0.f};
        const bf16* bp = bstage + mrow * BSTRIDE;
        #pragma unroll
        for (int ks = 0; ks < 16; ++ks) {
            const int k0 = ks * 32;
            v16bf bfr;
            #pragma unroll
            for (int j = 0; j < 8; ++j) {
                bfr[j]     = bp[k0 + kb1 + j];
                bfr[8 + j] = bp[k0 + kb1 + 16 + j];
            }
            acc = __builtin_amdgcn_wmma_f32_16x16x32_bf16(
                      false, afrag[ks], false, bfr, (short)0, acc, false, false);
        }
        const float bia = biasS[nt * 16 + mrow];
        #pragma unroll
        for (int r = 0; r < 8; ++r) {
            const int row = r + ((lane < 16) ? 0 : 8);      // C-matrix layout
            sc[wave][row][nt * 16 + mrow] = (acc[r] + bia) * 10.0f;  // 1/TEMP
        }
    }

    // ================= softmax (per wave: 2 lanes per row) ==================
    __syncthreads();
    {
        const int half = lane >> 4;
        v4f* rv = (v4f*)(&sc[wave][mrow][0] + half * 256);
        float mx = -1e30f;
        #pragma unroll 4
        for (int i = 0; i < 64; ++i) {
            v4f v = rv[i];
            mx = fmaxf(mx, fmaxf(fmaxf(v[0], v[1]), fmaxf(v[2], v[3])));
        }
        mx = fmaxf(mx, __shfl_xor(mx, 16, 32));
        float sum = 0.f;
        #pragma unroll 4
        for (int i = 0; i < 64; ++i) {
            v4f v = rv[i];
            v[0] = __expf(v[0] - mx); v[1] = __expf(v[1] - mx);
            v[2] = __expf(v[2] - mx); v[3] = __expf(v[3] - mx);
            sum += (v[0] + v[1]) + (v[2] + v[3]);
            rv[i] = v;
        }
        sum += __shfl_xor(sum, 16, 32);
        const float inv = 1.0f / sum;
        #pragma unroll 4
        for (int i = 0; i < 64; ++i) {
            v4f v = rv[i];
            v[0] *= inv; v[1] *= inv; v[2] *= inv; v[3] *= inv;
            rv[i] = v;
        }
    }

    // ---- write Q to HBM (coalesced 512B bursts, non-temporal: write-once)
    for (int r = 0; r < 16; ++r) {
        const v4f* src = (const v4f*)&sc[wave][r][0];
        v4f* dst = (v4f*)(Qout + (size_t)(m0 + r) * NDIM);
        #pragma unroll
        for (int i = 0; i < 4; ++i)
            __builtin_nontemporal_store(src[lane + i * 32], dst + lane + i * 32);
    }

    // ---- re-fragment Q (bf16) as A for GEMM-2, resident in VGPRs
    v16bf qfrag[16];
    {
        const float* qp = &sc[wave][mrow][0];
        #pragma unroll
        for (int ks = 0; ks < 16; ++ks) {
            const int k0 = ks * 32;
            v16bf f;
            #pragma unroll
            for (int j = 0; j < 8; ++j) {
                f[j]     = (bf16)qp[k0 + kb1 + j];
                f[8 + j] = (bf16)qp[k0 + kb1 + 16 + j];
            }
            qfrag[ks] = f;
        }
    }

    // ================= Phase C: Pt = Q @ Wp over 16 column tiles ============
    for (int ct = 0; ct < CDIM / 16; ++ct) {
        __syncthreads();
        {
            const uint4* src = (const uint4*)(WpT + ((size_t)ct * 16 + stgRow) * NDIM);
            uint4* dst = (uint4*)(bstage + stgRow * BSTRIDE);
            #pragma unroll
            for (int i = 0; i < 8; ++i) dst[stgSub + i * 8] = src[stgSub + i * 8];
        }
        __syncthreads();

        v8f acc = {0.f,0.f,0.f,0.f,0.f,0.f,0.f,0.f};
        const bf16* bp = bstage + mrow * BSTRIDE;
        #pragma unroll
        for (int ks = 0; ks < 16; ++ks) {
            const int k0 = ks * 32;
            v16bf bfr;
            #pragma unroll
            for (int j = 0; j < 8; ++j) {
                bfr[j]     = bp[k0 + kb1 + j];
                bfr[8 + j] = bp[k0 + kb1 + 16 + j];
            }
            acc = __builtin_amdgcn_wmma_f32_16x16x32_bf16(
                      false, qfrag[ks], false, bfr, (short)0, acc, false, false);
        }
        #pragma unroll
        for (int r = 0; r < 8; ++r) {
            const int row = r + ((lane < 16) ? 0 : 8);
            __builtin_nontemporal_store(
                acc[r], &Pt[(size_t)(m0 + row) * CDIM + ct * 16 + mrow]);
        }
    }
}

// ---------------------------------------------------------------------------
extern "C" void kernel_launch(void* const* d_in, const int* in_sizes, int n_in,
                              void* d_out, int out_size, void* d_ws, size_t ws_size,
                              hipStream_t stream)
{
    const float* Hl  = (const float*)d_in[0];   // (B,T,V,H)
    const float* Wp  = (const float*)d_in[1];   // (N,C)
    const float* Wq  = (const float*)d_in[2];   // (C,H)
    const float* Wqb = (const float*)d_in[3];   // (C)

    bf16*  Ws  = (bf16*)d_ws;                                               // 512 KB
    bf16*  WpT = (bf16*)((char*)d_ws + (size_t)NDIM * HDIM * 2);            // 256 KB
    float* bia = (float*)((char*)d_ws + (size_t)NDIM * HDIM * 2
                                      + (size_t)CDIM * NDIM * 2);           // 2 KB

    float* Pt   = (float*)d_out;                   // (M, C) first
    float* Qout = Pt + (size_t)MTOT * CDIM;        // then (M, N)

    hipLaunchKernelGGL(prep_kernel, dim3(NDIM), dim3(128), 0, stream,
                       Wp, Wq, Wqb, Ws, WpT, bia);
    hipLaunchKernelGGL(pq_fused_kernel, dim3(MTOT / ROWS_PER_BLOCK), dim3(TPB), 0, stream,
                       Hl, Ws, WpT, bia, Pt, Qout);
}
